// AxialCrossAttention_23648089932294
// MI455X (gfx1250) — matrix operand order repaired and checked by reading, not compiled, not run on hardware
//
#include <hip/hip_runtime.h>

typedef __bf16 bf16_t;
typedef __attribute__((ext_vector_type(16))) __bf16 v16bf;
typedef __attribute__((ext_vector_type(8)))  __bf16 v8bf;
typedef __attribute__((ext_vector_type(8)))  float  v8f;

struct U8x32 { uint4 a, b; };   // 32 bytes

__device__ __forceinline__ bf16_t f2bf(float f) {
  unsigned u = __builtin_bit_cast(unsigned, f);
  unsigned r = (u + 0x7FFFu + ((u >> 16) & 1u)) >> 16;
  unsigned short s = (unsigned short)r;
  return __builtin_bit_cast(bf16_t, s);
}
__device__ __forceinline__ float bf2f(bf16_t b) {
  unsigned short s = __builtin_bit_cast(unsigned short, b);
  unsigned u = ((unsigned)s) << 16;
  return __builtin_bit_cast(float, u);
}
__device__ __forceinline__ v16bf load16(const bf16_t* p) {
  U8x32 u = *(const U8x32*)p;
  return __builtin_bit_cast(v16bf, u);
}
__device__ __forceinline__ v16bf load8x2(const bf16_t* p0, const bf16_t* p1) {
  v8bf lo = __builtin_bit_cast(v8bf, *(const uint4*)p0);
  v8bf hi = __builtin_bit_cast(v8bf, *(const uint4*)p1);
  return __builtin_shufflevector(lo, hi, 0,1,2,3,4,5,6,7,8,9,10,11,12,13,14,15);
}
__device__ __forceinline__ v8f zero8() {
  v8f z = {0.f,0.f,0.f,0.f,0.f,0.f,0.f,0.f};
  return z;
}
#define WMMA_BF16(A_, B_, C_) \
  __builtin_amdgcn_wmma_f32_16x16x32_bf16(false, (A_), false, (B_), (short)0, (C_), false, false)

// ---------------- Tensor Data Mover (TDM) staging, probe-guarded ----------------
#if defined(__has_builtin)
# if __has_builtin(__builtin_amdgcn_tensor_load_to_lds) && __has_builtin(__builtin_amdgcn_s_wait_tensorcnt)
#  define HAVE_TDM 1
# endif
#endif
#ifndef HAVE_TDM
# define HAVE_TDM 0
#endif

#if HAVE_TDM
typedef __attribute__((ext_vector_type(4))) unsigned int tdm_v4u;
typedef __attribute__((ext_vector_type(8))) int tdm_v8i;
typedef __attribute__((ext_vector_type(4))) int tdm_v4i;

// 1-D contiguous global->LDS DMA of `elems` 2-byte elements (elems < 65536 so it
// fits tile_dim0). D# built per CDNA5 ISA 8.3/8.4: group0 = {count, lds_addr,
// global_addr, type=2}; group1 = {data_size=2B, tensor_dim0=tile_dim0=elems,
// tensor_dim1=tile_dim1=1, dim0_stride=elems}. Groups 2/3 zero (<=2D tensor).
__device__ __forceinline__ void tdm_load_1d(void* lds_dst, const void* gsrc, unsigned elems) {
  unsigned lds_off = (unsigned)(unsigned long long)lds_dst;   // addr[31:0] == LDS offset
  unsigned long long ga = (unsigned long long)gsrc;
  tdm_v4u g0;
  g0[0] = 1u;                                                 // count=1, user mode
  g0[1] = lds_off;
  g0[2] = (unsigned)ga;
  g0[3] = (unsigned)((ga >> 32) & 0x01FFFFFFull) | (2u << 30); // global_addr[56:32], type=2
  tdm_v8i g1;
  g1[0] = 0x10000;                                            // data_size=1 -> 2 bytes
  g1[1] = (int)((elems & 0xFFFFu) << 16);                     // tensor_dim0 lo16
  g1[2] = (int)((elems >> 16) & 0xFFFFu) | (1 << 16);         // tensor_dim0 hi16, tensor_dim1=1
  g1[3] = (int)((elems & 0xFFFFu) << 16);                     // tile_dim0
  g1[4] = 1;                                                  // tile_dim1=1, tile_dim2=0
  g1[5] = (int)elems;                                         // tensor_dim0_stride lo32
  g1[6] = 0;
  g1[7] = 0;
  tdm_v4i gz = {0, 0, 0, 0};
#if __clang_major__ >= 23
  tdm_v8i gz8 = {0, 0, 0, 0, 0, 0, 0, 0};
  __builtin_amdgcn_tensor_load_to_lds(g0, g1, gz, gz, gz8, 0);
#else
  __builtin_amdgcn_tensor_load_to_lds(g0, g1, gz, gz, 0);
#endif
}
#endif

// ---------------- weight prep ----------------
// dst[n*dst_ld + k] = src[k*src_ld + src_off + n]   (transpose+convert)
__global__ void wtr_kernel(const float* __restrict__ src, bf16_t* __restrict__ dst,
                           int K, int N, int src_ld, int src_off, int dst_ld) {
  int i = blockIdx.x * blockDim.x + threadIdx.x;
  if (i >= K * N) return;
  int n = i / K, k = i - n * K;
  dst[(size_t)n * dst_ld + k] = f2bf(src[(size_t)k * src_ld + src_off + n]);
}
__global__ void wcvt_kernel(const float* __restrict__ src, bf16_t* __restrict__ dst, int total) {
  int i = blockIdx.x * blockDim.x + threadIdx.x;
  if (i < total) dst[i] = f2bf(src[i]);
}
__global__ void bnprep_kernel(const float* g, const float* beta, const float* mu, const float* var,
                              const float* b1, const float* b2,
                              float* bnscale, float* bsum) {
  int n = threadIdx.x;  // 256 threads
  float s = g[n] * rsqrtf(var[n] + 1e-5f);
  bnscale[n] = s;
  bsum[n] = beta[n] - mu[n] * s + b1[n] + b2[n];
}

// ---------------- layernorm (NCHW -> token-major bf16, two affine variants) ----------------
__global__ __launch_bounds__(256)
void ln_kernel(const float* __restrict__ x,
               const float* __restrict__ wq, const float* __restrict__ bq,
               const float* __restrict__ wkv, const float* __restrict__ bkv,
               bf16_t* __restrict__ aq, bf16_t* __restrict__ akv, bf16_t* __restrict__ araw,
               int C) {
  __shared__ bf16_t xs[64 * 384];     // [c][w]
  __shared__ float mu[64], rs[64];
  int blk = blockIdx.x;               // b*64 + h
  int b = blk >> 6, h = blk & 63;
  size_t tbase = (size_t)blk * 64;    // token base
  const float* xin = x + (size_t)b * C * 4096 + h * 64;
  for (int e = threadIdx.x; e < 64 * C; e += 256) {
    int c = e >> 6, w = e & 63;
    xs[c * 64 + w] = f2bf(xin[(size_t)c * 4096 + w]);
  }
  __syncthreads();
  if (threadIdx.x < 64) {
    int w = threadIdx.x;
    float s = 0.f, ss = 0.f;
    for (int c = 0; c < C; ++c) { float v = bf2f(xs[c * 64 + w]); s += v; ss += v * v; }
    float mean = s / C;
    float varr = ss / C - mean * mean;
    mu[w] = mean; rs[w] = rsqrtf(varr + 1e-5f);
  }
  __syncthreads();
  for (int e = threadIdx.x; e < 64 * C; e += 256) {
    int w = e / C, c = e - w * C;
    bf16_t raw = xs[c * 64 + w];
    float v = (bf2f(raw) - mu[w]) * rs[w];
    size_t oi = (tbase + w) * C + c;
    aq[oi]  = f2bf(v * wq[c]  + bq[c]);
    akv[oi] = f2bf(v * wkv[c] + bkv[c]);
    if (araw) araw[oi] = raw;
  }
}

// ---------------- generic 64-token-tile GEMM ----------------
// A: [M][K] bf16 token-major; Bt: [N][K] bf16 (weights transposed)
// mode 0: out0[token][n] (ld = N)   mode 1: scatter into vtr/vtc transposed V layouts
__global__ __launch_bounds__(256)
void gemm64_kernel(const bf16_t* __restrict__ A, const bf16_t* __restrict__ Bt,
                   bf16_t* __restrict__ out0, bf16_t* __restrict__ vtr, bf16_t* __restrict__ vtc,
                   int K, int N, int mode) {
  __shared__ bf16_t As[64 * 384];
  int t0 = blockIdx.x * 64;
#if HAVE_TDM
  if (threadIdx.x < 32) {          // wave 0 issues one TDM op for the whole A tile
    tdm_load_1d(As, A + (size_t)t0 * K, (unsigned)(64 * K));
    __builtin_amdgcn_s_wait_tensorcnt(0);
  }
#else
  {
    const uint4* src = (const uint4*)(A + (size_t)t0 * K);
    uint4* dst = (uint4*)As;
    int total = (64 * K) >> 3;
    for (int i = threadIdx.x; i < total; i += 256) dst[i] = src[i];
  }
#endif
  __syncthreads();
  int wave = threadIdx.x >> 5, lane = threadIdx.x & 31;
  int m = lane & 15, half = lane >> 4;
  int ntiles = N >> 4;
  for (int tile = wave; tile < 4 * ntiles; tile += 8) {
    int mt = tile / ntiles, nt = tile - mt * ntiles;
    v8f acc = zero8();
    const bf16_t* arow = &As[(mt * 16 + m) * K];
    const bf16_t* brow = Bt + (size_t)(nt * 16 + m) * K + half * 16;
    for (int kc = 0; kc < K; kc += 32) {
      v16bf af = load16(arow + kc + half * 16);
      v16bf bfv = load16(brow + kc);
      acc = WMMA_BF16(af, bfv, acc);
    }
    if (mode == 0) {
      #pragma unroll
      for (int r = 0; r < 8; ++r) {
        int row = mt * 16 + r + 8 * half;
        out0[(size_t)(t0 + row) * N + nt * 16 + m] = f2bf(acc[r]);
      }
    } else {
      int b = blockIdx.x >> 6, h = blockIdx.x & 63;   // block row = (b,h), tokens over w
      #pragma unroll
      for (int r = 0; r < 8; ++r) {
        int w = mt * 16 + r + 8 * half;
        int col = nt * 16 + m;
        bf16_t v = f2bf(acc[r]);
        vtr[(((size_t)(b * 64 + w) * 256) + col) * 64 + h] = v;   // [B][w][d][h]
        vtc[(((size_t)(b * 64 + h) * 256) + col) * 64 + w] = v;   // [B][h][d][w]
      }
    }
  }
}

// ---------------- axial attention: one wave = one head, 64x64 scores ----------------
// Q,Kb: [B*4096][256] bf16 ; Vt: [B][64(pos)][256(d)][64(key)] bf16 ; O: [B*4096][256]
__global__ __launch_bounds__(256)
void attn_kernel(const bf16_t* __restrict__ Q, const bf16_t* __restrict__ Kb,
                 const bf16_t* __restrict__ Vt, bf16_t* __restrict__ O,
                 int axis, int accum) {
  __shared__ bf16_t P[8][16 * 64];
  int b = blockIdx.x >> 6, pos = blockIdx.x & 63;
  int wave = threadIdx.x >> 5, lane = threadIdx.x & 31;
  int head = wave;
  int m = lane & 15, half = lane >> 4;
  const float scale = 0.17677669529663687f;   // 1/sqrt(32)
  for (int rb = 0; rb < 4; ++rb) {
    // --- S = Q K^T ---
    int qi = rb * 16 + m;
    int t = (axis == 0) ? ((b * 64 + qi) * 64 + pos) : ((b * 64 + pos) * 64 + qi);
    const bf16_t* qp = Q + (size_t)t * 256 + head * 32;
    v16bf aq = load16(qp + half * 16);
    v8f s[4];
    #pragma unroll
    for (int kb = 0; kb < 4; ++kb) {
      int key = kb * 16 + m;
      int kt = (axis == 0) ? ((b * 64 + key) * 64 + pos) : ((b * 64 + pos) * 64 + key);
      v16bf bk = load16(Kb + (size_t)kt * 256 + head * 32 + half * 16);
      s[kb] = WMMA_BF16(aq, bk, zero8());
    }
    // --- softmax over 64 keys (per C-layout row; half-wave reductions) ---
    #pragma unroll
    for (int r = 0; r < 8; ++r) {
      float v0 = s[0][r] * scale, v1 = s[1][r] * scale, v2 = s[2][r] * scale, v3 = s[3][r] * scale;
      float mx = fmaxf(fmaxf(v0, v1), fmaxf(v2, v3));
      for (int d = 1; d < 16; d <<= 1) mx = fmaxf(mx, __shfl_xor(mx, d, 32));
      float e0 = __expf(v0 - mx), e1 = __expf(v1 - mx), e2 = __expf(v2 - mx), e3 = __expf(v3 - mx);
      float sm = e0 + e1 + e2 + e3;
      for (int d = 1; d < 16; d <<= 1) sm += __shfl_xor(sm, d, 32);
      float inv = 1.0f / sm;
      int row = r + 8 * half;
      bf16_t* pp = &P[wave][row * 64 + m];
      pp[0] = f2bf(e0 * inv); pp[16] = f2bf(e1 * inv);
      pp[32] = f2bf(e2 * inv); pp[48] = f2bf(e3 * inv);
    }
    __syncthreads();
    // --- O = P V ---
    v8f o0 = zero8(), o1 = zero8();
    #pragma unroll
    for (int c = 0; c < 2; ++c) {
      const bf16_t* prow = &P[wave][m * 64 + c * 32];
      v16bf ap = load16(prow + half * 16);
      #pragma unroll
      for (int nt = 0; nt < 2; ++nt) {
        int d = nt * 16 + m;
        v16bf bv = load16(Vt + (((size_t)(b * 64 + pos) * 256) + head * 32 + d) * 64 + c * 32 + half * 16);
        if (nt == 0) o0 = WMMA_BF16(ap, bv, o0);
        else         o1 = WMMA_BF16(ap, bv, o1);
      }
    }
    #pragma unroll
    for (int r = 0; r < 8; ++r) {
      int rl = rb * 16 + r + 8 * half;
      int tt = (axis == 0) ? ((b * 64 + rl) * 64 + pos) : ((b * 64 + pos) * 64 + rl);
      bf16_t* op = O + (size_t)tt * 256 + head * 32 + m;
      float v0 = o0[r], v1 = o1[r];
      if (accum) { v0 += bf2f(op[0]); v1 += bf2f(op[16]); }
      op[0] = f2bf(v0); op[16] = f2bf(v1);
    }
    __syncthreads();
  }
}

// ---------------- fused output: [o1|o2] @ [Wout1^T|Wout2^T] + BN(sc) + x2 + NCHW store ----------------
__global__ __launch_bounds__(256)
void fuse_kernel(const bf16_t* __restrict__ o1, const bf16_t* __restrict__ o2,
                 const bf16_t* __restrict__ Woutt /*[256][512]*/,
                 const bf16_t* __restrict__ scb,
                 const float* __restrict__ bnscale, const float* __restrict__ bsum,
                 const float* __restrict__ x2, float* __restrict__ out) {
  __shared__ bf16_t As1[64 * 256];
  __shared__ bf16_t As2[64 * 256];
  int t0 = blockIdx.x * 64;
#if HAVE_TDM
  if (threadIdx.x < 32) {          // wave 0 issues both TDM tile loads
    tdm_load_1d(As1, o1 + (size_t)t0 * 256, 64 * 256);
    tdm_load_1d(As2, o2 + (size_t)t0 * 256, 64 * 256);
    __builtin_amdgcn_s_wait_tensorcnt(0);
  }
#else
  {
    uint4* d1 = (uint4*)As1;
    uint4* d2 = (uint4*)As2;
    const uint4* s1 = (const uint4*)(o1 + (size_t)t0 * 256);
    const uint4* s2 = (const uint4*)(o2 + (size_t)t0 * 256);
    for (int i = threadIdx.x; i < 64 * 32; i += 256) { d1[i] = s1[i]; d2[i] = s2[i]; }
  }
#endif
  __syncthreads();
  int wave = threadIdx.x >> 5, lane = threadIdx.x & 31;
  int m = lane & 15, half = lane >> 4;
  for (int tile = wave; tile < 64; tile += 8) {
    int mt = tile >> 4, nt = tile & 15;
    v8f acc = zero8();
    const bf16_t* arow1 = &As1[(mt * 16 + m) * 256];
    const bf16_t* arow2 = &As2[(mt * 16 + m) * 256];
    const bf16_t* brow = Woutt + (size_t)(nt * 16 + m) * 512 + half * 16;
    #pragma unroll 4
    for (int kc = 0; kc < 256; kc += 32) {
      v16bf af = load16(arow1 + kc + half * 16);
      v16bf bfv = load16(brow + kc);
      acc = WMMA_BF16(af, bfv, acc);
    }
    #pragma unroll 4
    for (int kc = 0; kc < 256; kc += 32) {
      v16bf af = load16(arow2 + kc + half * 16);
      v16bf bfv = load16(brow + 256 + kc);
      acc = WMMA_BF16(af, bfv, acc);
    }
    #pragma unroll
    for (int r = 0; r < 8; ++r) {
      int row = mt * 16 + r + 8 * half;
      int tt = t0 + row;
      int n = nt * 16 + m;
      int bb = tt >> 12, rem = tt & 4095;
      size_t oi = ((size_t)(bb * 256 + n) << 12) + rem;
      float val = acc[r] + bsum[n] + bnscale[n] * bf2f(scb[(size_t)tt * 256 + n]) + x2[oi];
      out[oi] = val;
    }
  }
}

// ---------------- host ----------------
extern "C" void kernel_launch(void* const* d_in, const int* in_sizes, int n_in,
                              void* d_out, int out_size, void* d_ws, size_t ws_size,
                              hipStream_t stream) {
  (void)in_sizes; (void)n_in; (void)out_size; (void)ws_size;
  const float* x1      = (const float*)d_in[0];
  const float* x2      = (const float*)d_in[1];
  const float* ln1q_w  = (const float*)d_in[2];
  const float* ln1q_b  = (const float*)d_in[3];
  const float* ln2kv_w = (const float*)d_in[4];
  const float* ln2kv_b = (const float*)d_in[5];
  const float* Wq1     = (const float*)d_in[6];
  const float* Wkv2    = (const float*)d_in[7];
  const float* Wout1   = (const float*)d_in[8];
  const float* bout1   = (const float*)d_in[9];
  const float* ln2q_w  = (const float*)d_in[10];
  const float* ln2q_b  = (const float*)d_in[11];
  const float* ln1kv_w = (const float*)d_in[12];
  const float* ln1kv_b = (const float*)d_in[13];
  const float* Wq2     = (const float*)d_in[14];
  const float* Wkv1    = (const float*)d_in[15];
  const float* Wout2   = (const float*)d_in[16];
  const float* bout2   = (const float*)d_in[17];
  const float* Wsc     = (const float*)d_in[18];
  const float* bn_g    = (const float*)d_in[19];
  const float* bn_b    = (const float*)d_in[20];
  const float* bn_m    = (const float*)d_in[21];
  const float* bn_v    = (const float*)d_in[22];

  const size_t M = 32768;
  char* ws = (char*)d_ws;
  size_t off = 0;
  auto alloc = [&](size_t bytes) -> char* {
    char* p = ws + off; off += (bytes + 255) & ~(size_t)255; return p;
  };
  bf16_t* aq1   = (bf16_t*)alloc(M * 384 * 2);
  bf16_t* akv1  = (bf16_t*)alloc(M * 384 * 2);
  bf16_t* araw1 = (bf16_t*)alloc(M * 384 * 2);
  bf16_t* aq2   = (bf16_t*)alloc(M * 256 * 2);
  bf16_t* akv2  = (bf16_t*)alloc(M * 256 * 2);
  bf16_t* q1b   = (bf16_t*)alloc(M * 256 * 2);
  bf16_t* k2b   = (bf16_t*)alloc(M * 256 * 2);
  bf16_t* q2b   = (bf16_t*)alloc(M * 256 * 2);
  bf16_t* k1b   = (bf16_t*)alloc(M * 256 * 2);
  bf16_t* v2tr  = (bf16_t*)alloc(M * 256 * 2);
  bf16_t* v2tc  = (bf16_t*)alloc(M * 256 * 2);
  bf16_t* v1tr  = (bf16_t*)alloc(M * 256 * 2);
  bf16_t* v1tc  = (bf16_t*)alloc(M * 256 * 2);
  bf16_t* scb   = (bf16_t*)alloc(M * 256 * 2);
  bf16_t* o1b   = (bf16_t*)alloc(M * 256 * 2);
  bf16_t* o2b   = (bf16_t*)alloc(M * 256 * 2);
  bf16_t* Wq1t  = (bf16_t*)alloc(256 * 384 * 2);
  bf16_t* Wk2t  = (bf16_t*)alloc(256 * 256 * 2);
  bf16_t* Wv2t  = (bf16_t*)alloc(256 * 256 * 2);
  bf16_t* Wq2t  = (bf16_t*)alloc(256 * 256 * 2);
  bf16_t* Wk1t  = (bf16_t*)alloc(256 * 384 * 2);
  bf16_t* Wv1t  = (bf16_t*)alloc(256 * 384 * 2);
  bf16_t* Wsct  = (bf16_t*)alloc(256 * 384 * 2);
  bf16_t* Woutt = (bf16_t*)alloc(256 * 512 * 2);
  float*  bnsc  = (float*)alloc(256 * 4);
  float*  bsum  = (float*)alloc(256 * 4);

  dim3 blk(256);
  // weight prep
  wtr_kernel<<<(384 * 256 + 255) / 256, blk, 0, stream>>>(Wq1,  Wq1t,       384, 256, 256, 0,   384);
  wtr_kernel<<<(256 * 256 + 255) / 256, blk, 0, stream>>>(Wkv2, Wk2t,       256, 256, 512, 0,   256);
  wtr_kernel<<<(256 * 256 + 255) / 256, blk, 0, stream>>>(Wkv2, Wv2t,       256, 256, 512, 256, 256);
  wtr_kernel<<<(256 * 256 + 255) / 256, blk, 0, stream>>>(Wq2,  Wq2t,       256, 256, 256, 0,   256);
  wtr_kernel<<<(384 * 256 + 255) / 256, blk, 0, stream>>>(Wkv1, Wk1t,       384, 256, 512, 0,   384);
  wtr_kernel<<<(384 * 256 + 255) / 256, blk, 0, stream>>>(Wkv1, Wv1t,       384, 256, 512, 256, 384);
  wtr_kernel<<<(256 * 256 + 255) / 256, blk, 0, stream>>>(Wout1, Woutt,     256, 256, 256, 0,   512);
  wtr_kernel<<<(256 * 256 + 255) / 256, blk, 0, stream>>>(Wout2, Woutt + 256, 256, 256, 256, 0, 512);
  wcvt_kernel<<<(256 * 384 + 255) / 256, blk, 0, stream>>>(Wsc, Wsct, 256 * 384);
  bnprep_kernel<<<1, 256, 0, stream>>>(bn_g, bn_b, bn_m, bn_v, bout1, bout2, bnsc, bsum);

  // layernorms
  ln_kernel<<<512, blk, 0, stream>>>(x1, ln1q_w, ln1q_b, ln1kv_w, ln1kv_b, aq1, akv1, araw1, 384);
  ln_kernel<<<512, blk, 0, stream>>>(x2, ln2q_w, ln2q_b, ln2kv_w, ln2kv_b, aq2, akv2, nullptr, 256);

  // projections
  gemm64_kernel<<<512, blk, 0, stream>>>(aq1,   Wq1t, q1b, nullptr, nullptr, 384, 256, 0);
  gemm64_kernel<<<512, blk, 0, stream>>>(akv1,  Wk1t, k1b, nullptr, nullptr, 384, 256, 0);
  gemm64_kernel<<<512, blk, 0, stream>>>(akv1,  Wv1t, nullptr, v1tr, v1tc,   384, 256, 1);
  gemm64_kernel<<<512, blk, 0, stream>>>(araw1, Wsct, scb, nullptr, nullptr, 384, 256, 0);
  gemm64_kernel<<<512, blk, 0, stream>>>(aq2,   Wq2t, q2b, nullptr, nullptr, 256, 256, 0);
  gemm64_kernel<<<512, blk, 0, stream>>>(akv2,  Wk2t, k2b, nullptr, nullptr, 256, 256, 0);
  gemm64_kernel<<<512, blk, 0, stream>>>(akv2,  Wv2t, nullptr, v2tr, v2tc,   256, 256, 1);

  // axial attention (row over H, col over W); col pass accumulates
  attn_kernel<<<512, blk, 0, stream>>>(q1b, k2b, v2tr, o1b, 0, 0);
  attn_kernel<<<512, blk, 0, stream>>>(q1b, k2b, v2tc, o1b, 1, 1);
  attn_kernel<<<512, blk, 0, stream>>>(q2b, k1b, v1tr, o2b, 0, 0);
  attn_kernel<<<512, blk, 0, stream>>>(q2b, k1b, v1tc, o2b, 1, 1);

  // fused output projection + shortcut + residual + transpose
  fuse_kernel<<<512, blk, 0, stream>>>(o1b, o2b, Woutt, scb, bnsc, bsum, x2, (float*)d_out);
}
